// BTT_24704651887066
// MI455X (gfx1250) — matrix operand order repaired and checked by reading, not compiled, not run on hardware
//
#include <hip/hip_runtime.h>

typedef __attribute__((ext_vector_type(16))) __bf16       v16bf;
typedef __attribute__((ext_vector_type(8)))  float        v8f;
typedef __attribute__((ext_vector_type(4)))  float        f32x4;
typedef __attribute__((ext_vector_type(4)))  unsigned int u32x4;

#define TTSIZE 4096

__device__ __forceinline__ unsigned short bf16_rne(float f) {
    unsigned int u = __float_as_uint(f);
    u += 0x7fffu + ((u >> 16) & 1u);
    return (unsigned short)(u >> 16);
}
__device__ __forceinline__ float bf16_to_f32(unsigned short h) {
    return __uint_as_float(((unsigned int)h) << 16);
}
__device__ __forceinline__ void split_bf16(float f, unsigned short& hi, unsigned short& lo) {
    hi = bf16_rne(f);
    lo = bf16_rne(f - bf16_to_f32(hi));
}

// ---------------------------------------------------------------------------
// Kernel 1: materialize G transposed (N-major):  Gt[(y*64+i)][(j*64+x)]
//           = sum_b core1[j,x,i,0,b] * core0[j,y,i,b,0],  stored bf16 hi/lo.
// ---------------------------------------------------------------------------
__global__ __launch_bounds__(256) void btt_build_g(
    const float* __restrict__ core0, const float* __restrict__ core1,
    unsigned short* __restrict__ Ghi, unsigned short* __restrict__ Glo)
{
    __shared__ __align__(16) float c1[64][16];   // c1[x][b]
    __shared__ __align__(16) float c0[64][16];   // c0[y][b]
    const int j = blockIdx.x >> 6;
    const int i = blockIdx.x & 63;
    const int t = threadIdx.x;
    {
        const int r  = t >> 2;          // 0..63
        const int bq = (t & 3) << 2;    // 0,4,8,12
        const size_t o = ((size_t)(j * 64 + r) * 64 + i) * 16 + bq;
        *(f32x4*)&c1[r][bq] = *(const f32x4*)(core1 + o);
        *(f32x4*)&c0[r][bq] = *(const f32x4*)(core0 + o);
    }
    __syncthreads();
    const int y  = t & 63;
    const int xb = (t >> 6) << 4;       // 0,16,32,48
    float c0r[16];
#pragma unroll
    for (int b = 0; b < 16; ++b) c0r[b] = c0[y][b];
    const size_t rowbase = (size_t)(y * 64 + i) * TTSIZE + (size_t)j * 64;
#pragma unroll
    for (int xo = 0; xo < 16; ++xo) {
        const int x = xb + xo;
        float acc = 0.f;
#pragma unroll
        for (int b = 0; b < 16; ++b) acc = fmaf(c1[x][b], c0r[b], acc);
        unsigned short hi, lo;
        split_bf16(acc, hi, lo);
        Ghi[rowbase + x] = hi;
        Glo[rowbase + x] = lo;
    }
}

// ---------------------------------------------------------------------------
// Kernel 2: pre-split X (fp32) into bf16 hi/lo arrays. Pure bandwidth,
// removes all conversion VALU from the GEMM hot loop (was redone 32x).
// ---------------------------------------------------------------------------
__global__ __launch_bounds__(256) void btt_split_x(
    const float* __restrict__ X,
    unsigned short* __restrict__ Xhi, unsigned short* __restrict__ Xlo)
{
    const size_t idx = ((size_t)blockIdx.x * 256 + threadIdx.x) * 8;
    f32x4 a = *(const f32x4*)(X + idx);
    f32x4 b = *(const f32x4*)(X + idx + 4);
    float f[8] = {a[0], a[1], a[2], a[3], b[0], b[1], b[2], b[3]};
    u32x4 hp, lp;
#pragma unroll
    for (int p = 0; p < 4; ++p) {
        unsigned short h0, l0, h1, l1;
        split_bf16(f[2 * p],     h0, l0);
        split_bf16(f[2 * p + 1], h1, l1);
        hp[p] = (unsigned int)h0 | ((unsigned int)h1 << 16);
        lp[p] = (unsigned int)l0 | ((unsigned int)l1 << 16);
    }
    *(u32x4*)(Xhi + idx) = hp;
    *(u32x4*)(Xlo + idx) = lp;
}

// ---------------------------------------------------------------------------
// Kernel 3: out[m,n] = sum_k X[m,k]*G[k,n] + bias[n]
// 128x128 tile / WG, BK=32, 8 waves (4m x 2n), 2x4 16x16 frags per wave.
// Split-bf16: 3x v_wmma_f32_16x16x32_bf16 per frag per k-step.
// PRESPLIT=true: A loaded from pre-split bf16 arrays (b128 copies only).
// PRESPLIT=false: fallback, split X in-kernel (small-ws path).
// ---------------------------------------------------------------------------
template <bool PRESPLIT>
__global__ __launch_bounds__(256) void btt_gemm(
    const float* __restrict__ X,
    const unsigned short* __restrict__ Xhi,
    const unsigned short* __restrict__ Xlo,
    const unsigned short* __restrict__ Ghi,
    const unsigned short* __restrict__ Glo,
    const float* __restrict__ bias,
    float* __restrict__ Out)
{
    __shared__ __align__(16) unsigned short sAhi[128 * 40];  // stride 40 halves:
    __shared__ __align__(16) unsigned short sAlo[128 * 40];  //  16B-aligned rows,
    __shared__ __align__(16) unsigned short sBhi[128 * 40];  //  bank-spread (gcd(20,64)=4)
    __shared__ __align__(16) unsigned short sBlo[128 * 40];

    const int tid = threadIdx.x;
    const int m0  = blockIdx.y * 128;
    const int n0  = blockIdx.x * 128;

    // loader mapping: each thread owns half a tile row (16 elements)
    const int lr = tid >> 1;           // row in tile 0..127
    const int lc = (tid & 1) << 4;     // 0 or 16
    const float*          xg  = X   + (size_t)(m0 + lr) * TTSIZE + lc;
    const unsigned short* xhg = Xhi + (size_t)(m0 + lr) * TTSIZE + lc;
    const unsigned short* xlg = Xlo + (size_t)(m0 + lr) * TTSIZE + lc;
    const unsigned short* gh  = Ghi + (size_t)(n0 + lr) * TTSIZE + lc;
    const unsigned short* gl  = Glo + (size_t)(n0 + lr) * TTSIZE + lc;

    const int lane  = tid & 31;
    const int wave  = tid >> 5;
    const int wm    = wave & 3;        // 0..3: 32-row slab
    const int wn    = wave >> 2;       // 0..1: 64-col slab
    const int mrow  = lane & 15;       // row/col within 16x16 frag
    const int khalf = lane >> 4;       // lane-half selects K split

    v8f acc[2][4];
#pragma unroll
    for (int s = 0; s < 2; ++s)
#pragma unroll
        for (int tt = 0; tt < 4; ++tt)
#pragma unroll
            for (int e = 0; e < 8; ++e) acc[s][tt][e] = 0.f;

    f32x4 xr[4];             // used only when !PRESPLIT
    u32x4 ahr[2], alr[2];    // used only when PRESPLIT
    u32x4 bhr[2], blr[2];

    // prime the pipeline with k=0 tile
    if (PRESPLIT) {
        ahr[0] = *(const u32x4*)(xhg);  ahr[1] = *(const u32x4*)(xhg + 8);
        alr[0] = *(const u32x4*)(xlg);  alr[1] = *(const u32x4*)(xlg + 8);
    } else {
#pragma unroll
        for (int q = 0; q < 4; ++q) xr[q] = *(const f32x4*)(xg + q * 4);
    }
    bhr[0] = *(const u32x4*)(gh);  bhr[1] = *(const u32x4*)(gh + 8);
    blr[0] = *(const u32x4*)(gl);  blr[1] = *(const u32x4*)(gl + 8);

    for (int kk = 0; kk < TTSIZE / 32; ++kk) {
        __syncthreads();  // previous compute done; LDS reusable
        {   // stage registers -> LDS
            const int ab = lr * 40 + lc;
            if (PRESPLIT) {
                *(u32x4*)&sAhi[ab]     = ahr[0];
                *(u32x4*)&sAhi[ab + 8] = ahr[1];
                *(u32x4*)&sAlo[ab]     = alr[0];
                *(u32x4*)&sAlo[ab + 8] = alr[1];
            } else {
#pragma unroll
                for (int q = 0; q < 4; ++q) {
#pragma unroll
                    for (int p = 0; p < 2; ++p) {
                        unsigned short h0, l0, h1, l1;
                        split_bf16(xr[q][2 * p],     h0, l0);
                        split_bf16(xr[q][2 * p + 1], h1, l1);
                        *(unsigned int*)&sAhi[ab + q * 4 + 2 * p] =
                            (unsigned int)h0 | ((unsigned int)h1 << 16);
                        *(unsigned int*)&sAlo[ab + q * 4 + 2 * p] =
                            (unsigned int)l0 | ((unsigned int)l1 << 16);
                    }
                }
            }
            const int bb = lr * 40 + lc;
            *(u32x4*)&sBhi[bb]     = bhr[0];
            *(u32x4*)&sBhi[bb + 8] = bhr[1];
            *(u32x4*)&sBlo[bb]     = blr[0];
            *(u32x4*)&sBlo[bb + 8] = blr[1];
        }
        __syncthreads();  // tile visible

        if (kk + 1 < TTSIZE / 32) {  // prefetch next tile (overlaps compute)
            const int k1 = (kk + 1) * 32;
            if (PRESPLIT) {
                ahr[0] = *(const u32x4*)(xhg + k1);  ahr[1] = *(const u32x4*)(xhg + k1 + 8);
                alr[0] = *(const u32x4*)(xlg + k1);  alr[1] = *(const u32x4*)(xlg + k1 + 8);
            } else {
#pragma unroll
                for (int q = 0; q < 4; ++q) xr[q] = *(const f32x4*)(xg + k1 + q * 4);
            }
            bhr[0] = *(const u32x4*)(gh + k1);  bhr[1] = *(const u32x4*)(gh + k1 + 8);
            blr[0] = *(const u32x4*)(gl + k1);  blr[1] = *(const u32x4*)(gl + k1 + 8);
        }

        // A fragments: lane half L<16 holds K {0..15}, L>=16 holds {16..31}
        union AF { unsigned int d[8]; v16bf v; } ah[2], al[2];
#pragma unroll
        for (int s = 0; s < 2; ++s) {
            const int rb = (wm * 32 + s * 16 + mrow) * 40;
#pragma unroll
            for (int v = 0; v < 4; ++v) {
                const int k0off = khalf * 16 + 2 * v;
                const int k1off = khalf * 16 + 8 + 2 * v;
                ah[s].d[v]     = *(const unsigned int*)&sAhi[rb + k0off];
                ah[s].d[4 + v] = *(const unsigned int*)&sAhi[rb + k1off];
                al[s].d[v]     = *(const unsigned int*)&sAlo[rb + k0off];
                al[s].d[4 + v] = *(const unsigned int*)&sAlo[rb + k1off];
            }
        }
        // B fragments: N = lane%16, K = khalf*16 + e (contiguous -> b128 loads)
        union BF { u32x4 q[2]; v16bf v; } bh[4], bl[4];
#pragma unroll
        for (int tt = 0; tt < 4; ++tt) {
            const int nb = (wn * 64 + tt * 16 + mrow) * 40 + khalf * 16;
            bh[tt].q[0] = *(const u32x4*)&sBhi[nb];
            bh[tt].q[1] = *(const u32x4*)&sBhi[nb + 8];
            bl[tt].q[0] = *(const u32x4*)&sBlo[nb];
            bl[tt].q[1] = *(const u32x4*)&sBlo[nb + 8];
        }
        // split-bf16 product: hi*hi + hi*lo + lo*hi
#pragma unroll
        for (int s = 0; s < 2; ++s) {
#pragma unroll
            for (int tt = 0; tt < 4; ++tt) {
                acc[s][tt] = __builtin_amdgcn_wmma_f32_16x16x32_bf16(
                    false, ah[s].v, false, bh[tt].v, (short)0, acc[s][tt], false, false);
                acc[s][tt] = __builtin_amdgcn_wmma_f32_16x16x32_bf16(
                    false, ah[s].v, false, bl[tt].v, (short)0, acc[s][tt], false, false);
                acc[s][tt] = __builtin_amdgcn_wmma_f32_16x16x32_bf16(
                    false, al[s].v, false, bh[tt].v, (short)0, acc[s][tt], false, false);
            }
        }
    }

    // epilogue: C frag layout M = v + 8*khalf, N = lane%16
#pragma unroll
    for (int tt = 0; tt < 4; ++tt) {
        const int gn = n0 + wn * 64 + tt * 16 + mrow;
        const float bv = bias[gn];
#pragma unroll
        for (int s = 0; s < 2; ++s) {
            const int gmb = m0 + wm * 32 + s * 16 + khalf * 8;
#pragma unroll
            for (int v = 0; v < 8; ++v) {
                Out[(size_t)(gmb + v) * TTSIZE + gn] = acc[s][tt][v] + bv;
            }
        }
    }
}

extern "C" void kernel_launch(void* const* d_in, const int* in_sizes, int n_in,
                              void* d_out, int out_size, void* d_ws, size_t ws_size,
                              hipStream_t stream) {
    const float* x     = (const float*)d_in[0];   // (2,4096,4096) fp32
    const float* core0 = (const float*)d_in[1];   // (64,64,64,16,1) fp32
    const float* core1 = (const float*)d_in[2];   // (64,64,64,1,16) fp32
    const float* bias  = (const float*)d_in[3];   // (4096,) fp32
    float* out = (float*)d_out;

    const int    M      = in_sizes[0] / TTSIZE;                 // 8192 rows
    const size_t gElems = (size_t)TTSIZE * TTSIZE;              // per G matrix
    const size_t xElems = (size_t)M * TTSIZE;                   // per X matrix
    const size_t needPresplit = (2 * gElems + 2 * xElems) * sizeof(unsigned short);

    unsigned short* Ghi = (unsigned short*)d_ws;    // 32 MB
    unsigned short* Glo = Ghi + gElems;             // 32 MB
    unsigned short* Xhi = Glo + gElems;             // 64 MB
    unsigned short* Xlo = Xhi + xElems;             // 64 MB

    const bool presplit = (ws_size >= needPresplit);  // fixed per run -> deterministic

    btt_build_g<<<dim3(64 * 64), dim3(256), 0, stream>>>(core0, core1, Ghi, Glo);
    if (presplit) {
        btt_split_x<<<dim3((unsigned int)(xElems / (256 * 8))), dim3(256), 0, stream>>>(x, Xhi, Xlo);
        btt_gemm<true><<<dim3(TTSIZE / 128, M / 128), dim3(256), 0, stream>>>(
            x, Xhi, Xlo, Ghi, Glo, bias, out);
    } else {
        btt_gemm<false><<<dim3(TTSIZE / 128, M / 128), dim3(256), 0, stream>>>(
            x, Ghi, Glo, Ghi, Glo, bias, out);  // Xhi/Xlo unused in this path
    }
}
